// UIModel_50405736186523
// MI455X (gfx1250) — matrix operand order repaired and checked by reading, not compiled, not run on hardware
//
#include <hip/hip_runtime.h>
#include <math.h>

// Model dims (fixed by the reference)
#define T_SEQ   2048
#define EMB_D   512
#define HID_D   1024
#define G4      4096     // 4*HID
#define VOCAB_N 50000
#define NOUT    5
#define MAXLEN  20
#define PAD_T   0
#define SOS_T   1
#define EOS_T   2

// Persistent-kernel geometry
#define NB      32       // blocks (must all be co-resident; 32 << WGP count)
#define BT      256      // threads per block = 8 waves

typedef _Float16 v8h  __attribute__((ext_vector_type(8)));
typedef _Float16 v16h __attribute__((ext_vector_type(16)));
typedef float    v8f  __attribute__((ext_vector_type(8)));

// ---- A-fragment (16x32 f16, MxK): lane r(0..15) holds row r.
// lanes 0-15: K = {k0+0..15}; lanes 16-31: K = {k0+16..31}.
// Caller passes p = rowbase + k0 + koffA (koffA = 0 or 16).
__device__ __forceinline__ v16h ldfragA(const _Float16* p) {
  v8h lo = *(const v8h*)(p);
  v8h hi = *(const v8h*)(p + 8);
  v16h r;
#pragma unroll
  for (int i = 0; i < 8; ++i) { r[i] = lo[i]; r[i + 8] = hi[i]; }
  return r;
}

__device__ __forceinline__ float sigf(float x) { return 1.0f / (1.0f + expf(-x)); }

// Device-wide barrier: monotonically increasing counter (reset by prep each launch).
__device__ __forceinline__ void grid_barrier(unsigned* cnt) {
  __builtin_amdgcn_fence(__ATOMIC_RELEASE, "agent");
  __syncthreads();
  if (threadIdx.x == 0) {
    unsigned old = __hip_atomic_fetch_add(cnt, 1u, __ATOMIC_RELAXED, __HIP_MEMORY_SCOPE_AGENT);
    unsigned target = (old / NB + 1u) * NB;
    while (__hip_atomic_load(cnt, __ATOMIC_RELAXED, __HIP_MEMORY_SCOPE_AGENT) < target)
      __builtin_amdgcn_s_sleep(2);
  }
  __syncthreads();
  __builtin_amdgcn_fence(__ATOMIC_ACQUIRE, "agent");
}

// ---------------------------------------------------------------------------
// Kernel 1: f32->f16 weight conversion + embedding gather + state init
// ---------------------------------------------------------------------------
__global__ void prep_kernel(const int* __restrict__ x, const float* __restrict__ embtab,
                            const float* __restrict__ eWih, const float* __restrict__ eWhh,
                            const float* __restrict__ dWih, const float* __restrict__ dWhh,
                            _Float16* WihE, _Float16* WhhE, _Float16* WihD, _Float16* WhhD,
                            _Float16* embseq, _Float16* embcur, _Float16* hbuf,
                            unsigned* ctrl) {
  const long N1 = (long)G4 * EMB_D;     // 2,097,152
  const long N2 = (long)G4 * HID_D;     // 4,194,304
  const long N5 = (long)T_SEQ * EMB_D;  // 1,048,576
  const long total = N1 + N2 + N1 + N2 + N5 + EMB_D + 2 * HID_D + 8;
  for (long i = (long)blockIdx.x * blockDim.x + threadIdx.x; i < total;
       i += (long)gridDim.x * blockDim.x) {
    long k = i;
    if (k < N1) { WihE[k] = (_Float16)eWih[k]; continue; } k -= N1;
    if (k < N2) { WhhE[k] = (_Float16)eWhh[k]; continue; } k -= N2;
    if (k < N1) { WihD[k] = (_Float16)dWih[k]; continue; } k -= N1;
    if (k < N2) { WhhD[k] = (_Float16)dWhh[k]; continue; } k -= N2;
    if (k < N5) {
      long t = k >> 9, e = k & (EMB_D - 1);
      int id = x[t];
      float v = (id == PAD_T) ? 0.0f : embtab[(long)id * EMB_D + e];
      embseq[k] = (_Float16)v; continue;
    } k -= N5;
    if (k < EMB_D) { embcur[k] = (_Float16)embtab[(long)SOS_T * EMB_D + k]; continue; } k -= EMB_D;
    if (k < 2 * HID_D) { hbuf[k] = (_Float16)0.0f; continue; } k -= 2 * HID_D;
    ctrl[k] = 0u;  // [0]=barrier counter, [1]=valid, [2]=done
  }
}

// ---------------------------------------------------------------------------
// Kernel 2: Xproj[T,4H] = embseq[T,E] @ WihE^T + (bih + bhh)   (f16 WMMA, f32 acc)
// One wave per 16x16 output tile; K-loop over E in steps of 32.
// ---------------------------------------------------------------------------
__global__ void __launch_bounds__(256) xproj_kernel(const _Float16* __restrict__ embseq,
                                                    const _Float16* __restrict__ WihE,
                                                    const float* __restrict__ bih,
                                                    const float* __restrict__ bhh,
                                                    float* __restrict__ Xp) {
  const int wave = (blockIdx.x * blockDim.x + threadIdx.x) >> 5;
  const int lane = threadIdx.x & 31;
  const int tile_m = wave >> 8;          // G4/16 = 256 tiles along N
  const int tile_n = wave & 255;
  const int t0 = tile_m * 16, n0 = tile_n * 16;
  const int r = lane & 15;
  const int koffA = (lane & 16) ? 16 : 0;
  const int koffB = (lane & 16) ? 16 : 0;
  const _Float16* arow = embseq + (long)(t0 + r) * EMB_D;
  const _Float16* brow = WihE + (long)(n0 + r) * EMB_D;   // B col n == Wih row n
  v8f c = {};
#pragma unroll 4
  for (int k0 = 0; k0 < EMB_D; k0 += 32) {
    v16h a = ldfragA(arow + k0 + koffA);
    v16h b = *(const v16h*)(brow + k0 + koffB);
    c = __builtin_amdgcn_wmma_f32_16x16x32_f16(false, a, false, b, (short)0, c, false, false);
  }
  const float bs = bih[n0 + r] + bhh[n0 + r];
  const int rb = t0 + ((lane & 16) ? 8 : 0);
#pragma unroll
  for (int i = 0; i < 8; ++i)
    Xp[(long)(rb + i) * G4 + n0 + r] = c[i] + bs;
}

// ---------------------------------------------------------------------------
// Kernel 3: persistent encoder recurrence + decoder.
// Block b owns hidden units [b*32, b*32+32); its 8 waves compute the matching
// i/f/g/o gate rows (16 per wave) with WMMA (h broadcast into all 16 A rows).
// h is double-buffered in global -> one grid barrier per encoder step.
// ---------------------------------------------------------------------------
__global__ void __launch_bounds__(BT, 1) rnn_kernel(
    const _Float16* __restrict__ WhhE,
    const _Float16* __restrict__ WihD,
    const _Float16* __restrict__ WhhD,
    const float* __restrict__ Xp,
    _Float16* hbuf, float* hnewf, _Float16* embcur, unsigned* ctrl,
    const float* __restrict__ dbih, const float* __restrict__ dbhh,
    const float* __restrict__ outW, const float* __restrict__ outB,
    const float* __restrict__ embtab, float* dout) {
  const int b = blockIdx.x;
  const int tid = threadIdx.x;
  const int w = tid >> 5, lane = tid & 31, r = lane & 15;
  const int q = w >> 1, sub = w & 1;                    // q = gate quadrant (i,f,g,o)
  const int g0 = q * HID_D + b * 32 + sub * 16;         // 16 gate rows for this wave
  const int koffA = (lane & 16) ? 16 : 0;
  const int koffB = (lane & 16) ? 16 : 0;

  __shared__ float lgate[4][32];
  __shared__ float spred[NOUT];
  __shared__ int snid, svalid;

  float cstate = 0.0f;          // cell state for unit j (tid < 32 only)
  int cur = 0;
  const int j = b * 32 + tid;   // owned hidden unit (tid < 32)

  // ---------------- encoder: 2048 sequential steps ----------------
  for (int t = 0; t < T_SEQ; ++t) {
    const _Float16* hrow = hbuf + cur * HID_D;
    const _Float16* brow = WhhE + (long)(g0 + r) * HID_D;
    v8f c = {};
    for (int k0 = 0; k0 < HID_D; k0 += 32) {
      v16h a = ldfragA(hrow + k0 + koffA);              // h broadcast to all 16 rows
      v16h bm = *(const v16h*)(brow + k0 + koffB);
      c = __builtin_amdgcn_wmma_f32_16x16x32_f16(false, a, false, bm, (short)0, c, false, false);
    }
    if (lane < 16)   // row M=0 of D lives in VGPR0, lanes 0-15
      lgate[q][sub * 16 + r] = c[0] + Xp[(long)t * G4 + g0 + r];
    __syncthreads();
    if (tid < 32) {
      float iv = lgate[0][tid], fv = lgate[1][tid], gv = lgate[2][tid], ov = lgate[3][tid];
      cstate = sigf(fv) * cstate + sigf(iv) * tanhf(gv);
      float hn = sigf(ov) * tanhf(cstate);
      hbuf[(cur ^ 1) * HID_D + j] = (_Float16)hn;       // write next buffer
    }
    grid_barrier(ctrl);
    cur ^= 1;
  }

  // ---------------- decoder: 20 sequential steps ----------------
  for (int s = 0; s < MAXLEN; ++s) {
    const _Float16* hrow = hbuf + cur * HID_D;
    const _Float16* be = WihD + (long)(g0 + r) * EMB_D;
    const _Float16* bh = WhhD + (long)(g0 + r) * HID_D;
    v8f c = {};
    for (int k0 = 0; k0 < EMB_D; k0 += 32) {            // emb @ WihD^T
      v16h a = ldfragA(embcur + k0 + koffA);
      v16h bm = *(const v16h*)(be + k0 + koffB);
      c = __builtin_amdgcn_wmma_f32_16x16x32_f16(false, a, false, bm, (short)0, c, false, false);
    }
    for (int k0 = 0; k0 < HID_D; k0 += 32) {            // h @ WhhD^T
      v16h a = ldfragA(hrow + k0 + koffA);
      v16h bm = *(const v16h*)(bh + k0 + koffB);
      c = __builtin_amdgcn_wmma_f32_16x16x32_f16(false, a, false, bm, (short)0, c, false, false);
    }
    if (lane < 16)
      lgate[q][sub * 16 + r] = c[0] + dbih[g0 + r] + dbhh[g0 + r];
    __syncthreads();
    float cnew = 0.0f, hn = 0.0f;
    if (tid < 32) {
      float iv = lgate[0][tid], fv = lgate[1][tid], gv = lgate[2][tid], ov = lgate[3][tid];
      cnew = sigf(fv) * cstate + sigf(iv) * tanhf(gv);
      hn = sigf(ov) * tanhf(cnew);
      hnewf[j] = hn;                                    // candidate h (f32) for out-proj
    }
    grid_barrier(ctrl);                                  // A: hnewf visible everywhere

    if (b == 0) {
      if (tid < NOUT) {                                  // pred = h_new @ outW^T + outB
        float acc = outB[tid];
        for (int k = 0; k < HID_D; ++k) acc += hnewf[k] * outW[tid * HID_D + k];
        spred[tid] = acc;
      }
      __syncthreads();
      if (tid == 0) {
        int done = (int)ctrl[2];
        int valid = !done;
        float rj = rintf(spred[0]);                      // round-half-even like jnp.round
        if (rj < 0.0f) rj = 0.0f;
        if (rj > (float)(VOCAB_N - 1)) rj = (float)(VOCAB_N - 1);
        int nid = (int)rj;
        snid = nid; svalid = valid;
        ctrl[1] = (unsigned)valid;
        ctrl[2] = (unsigned)(done || (valid && nid == EOS_T));
        dout[MAXLEN * NOUT + s] = valid ? 1.0f : 0.0f;   // valids output
      }
      __syncthreads();
      int valid = svalid, nid = snid;
      if (tid < NOUT) dout[s * NOUT + tid] = valid ? spred[tid] : 0.0f;
      if (valid) {                                       // next-token embedding
        for (int e = tid; e < EMB_D; e += BT) {
          float ev = (nid == PAD_T) ? 0.0f : embtab[(long)nid * EMB_D + e];
          embcur[e] = (_Float16)ev;
        }
      }
    }
    grid_barrier(ctrl);                                  // B: valid/embcur visible

    int valid = (int)__hip_atomic_load(&ctrl[1], __ATOMIC_RELAXED, __HIP_MEMORY_SCOPE_AGENT);
    if (tid < 32) {
      if (valid) cstate = cnew;
      float hold = valid ? hn : (float)hbuf[cur * HID_D + j];
      hbuf[(cur ^ 1) * HID_D + j] = (_Float16)hold;
    }
    grid_barrier(ctrl);                                  // C: committed h visible
    cur ^= 1;
  }
}

// ---------------------------------------------------------------------------
extern "C" void kernel_launch(void* const* d_in, const int* in_sizes, int n_in,
                              void* d_out, int out_size, void* d_ws, size_t ws_size,
                              hipStream_t stream) {
  (void)in_sizes; (void)n_in; (void)out_size; (void)ws_size;
  const int*   x       = (const int*)  d_in[0];
  const float* embtab  = (const float*)d_in[1];
  const float* eWih    = (const float*)d_in[2];
  const float* eWhh    = (const float*)d_in[3];
  const float* ebih    = (const float*)d_in[4];
  const float* ebhh    = (const float*)d_in[5];
  const float* dWih    = (const float*)d_in[6];
  const float* dWhh    = (const float*)d_in[7];
  const float* dbih    = (const float*)d_in[8];
  const float* dbhh    = (const float*)d_in[9];
  const float* outW    = (const float*)d_in[10];
  const float* outB    = (const float*)d_in[11];

  char* ws = (char*)d_ws;
  size_t off = 0;
  auto take = [&](size_t bytes) { char* p = ws + off; off += (bytes + 255) & ~(size_t)255; return p; };
  _Float16* WihE   = (_Float16*)take((size_t)G4 * EMB_D * 2);
  _Float16* WhhE   = (_Float16*)take((size_t)G4 * HID_D * 2);
  _Float16* WihD   = (_Float16*)take((size_t)G4 * EMB_D * 2);
  _Float16* WhhD   = (_Float16*)take((size_t)G4 * HID_D * 2);
  _Float16* embseq = (_Float16*)take((size_t)T_SEQ * EMB_D * 2);
  float*    Xproj  = (float*)   take((size_t)T_SEQ * G4 * 4);
  _Float16* hbuf   = (_Float16*)take((size_t)2 * HID_D * 2);
  float*    hnewf  = (float*)   take((size_t)HID_D * 4);
  _Float16* embcur = (_Float16*)take((size_t)EMB_D * 2);
  unsigned* ctrl   = (unsigned*)take(8 * sizeof(unsigned));

  prep_kernel<<<4096, 256, 0, stream>>>(x, embtab, eWih, eWhh, dWih, dWhh,
                                        WihE, WhhE, WihD, WhhD,
                                        embseq, embcur, hbuf, ctrl);
  // (T/16)*(G4/16) = 128*256 = 32768 wave-tiles, 8 waves/block -> 4096 blocks
  xproj_kernel<<<4096, 256, 0, stream>>>(embseq, WihE, ebih, ebhh, Xproj);
  rnn_kernel<<<NB, BT, 0, stream>>>(WhhE, WihD, WhhD, Xproj, hbuf, hnewf, embcur, ctrl,
                                    dbih, dbhh, outW, outB, embtab, (float*)d_out);
}
